// GlobalSpatialAttention_87333864997138
// MI455X (gfx1250) — matrix operand order repaired and checked by reading, compiled, not run on hardware
//
#include <hip/hip_runtime.h>

#define NB  4
#define CIN 256
#define CQ  64
#define HWD 4096

typedef __attribute__((ext_vector_type(16))) __bf16 v16bf;
typedef __attribute__((ext_vector_type(8)))  __bf16 v8bf;
typedef __attribute__((ext_vector_type(8)))  float  v8f;

#if __has_builtin(__builtin_amdgcn_global_load_async_to_lds_b128) && __has_builtin(__builtin_amdgcn_s_wait_asynccnt)
#define ASYNC_OK 1
#else
#define ASYNC_OK 0
#endif

#if __has_builtin(__builtin_amdgcn_tensor_load_to_lds) && __has_builtin(__builtin_amdgcn_s_wait_tensorcnt)
#define TDM_OK 1
#else
#define TDM_OK 0
#endif
#if TDM_OK && (__has_include(<hip/amd_detail/amd_gfx1250_TDM.h>) || (__clang_major__ >= 23))
#define TDM_SIX 1
#else
#define TDM_SIX 0
#endif

static __device__ inline v8f vzero8() {
  v8f z = {0.f, 0.f, 0.f, 0.f, 0.f, 0.f, 0.f, 0.f};
  return z;
}

static __device__ inline v8f wmma_bf16(v16bf a, v16bf b, v8f c) {
  return __builtin_amdgcn_wmma_f32_16x16x32_bf16(false, a, false, b, (short)0, c, false, false);
}

// A operand (16M x 32K bf16) from a row-major [rows][ld] bf16 matrix.
static __device__ inline v16bf load_A_rm(const __bf16* g, int ld, int row0, int k0) {
  int lane = (int)(threadIdx.x & 31u);
  int m = row0 + (lane & 15);
  int koff = (lane < 16) ? 0 : 8;
  const __bf16* p = g + (size_t)m * ld + k0 + koff;
  v8bf lo = *(const v8bf*)(p);
  v8bf hi = *(const v8bf*)(p + 16);
  return __builtin_shufflevector(lo, hi, 0,1,2,3,4,5,6,7,8,9,10,11,12,13,14,15);
}

// Same A layout but from f32 row-major (weights), converting to bf16.
static __device__ inline v16bf load_A_f32(const float* g, int ld, int row0, int k0) {
  int lane = (int)(threadIdx.x & 31u);
  int m = row0 + (lane & 15);
  int koff = (lane < 16) ? 0 : 8;
  const float* p = g + (size_t)m * ld + k0 + koff;
  float4 f0 = *(const float4*)(p + 0);
  float4 f1 = *(const float4*)(p + 4);
  float4 f2 = *(const float4*)(p + 16);
  float4 f3 = *(const float4*)(p + 20);
  v16bf a;
  a[0]=(__bf16)f0.x;  a[1]=(__bf16)f0.y;  a[2]=(__bf16)f0.z;  a[3]=(__bf16)f0.w;
  a[4]=(__bf16)f1.x;  a[5]=(__bf16)f1.y;  a[6]=(__bf16)f1.z;  a[7]=(__bf16)f1.w;
  a[8]=(__bf16)f2.x;  a[9]=(__bf16)f2.y;  a[10]=(__bf16)f2.z; a[11]=(__bf16)f2.w;
  a[12]=(__bf16)f3.x; a[13]=(__bf16)f3.y; a[14]=(__bf16)f3.z; a[15]=(__bf16)f3.w;
  return a;
}

// B operand (32K x 16N bf16) from a TRANSPOSED [n][ld] store.
static __device__ inline v16bf load_B_tr(const __bf16* t, int ld, int n0, int k0) {
  int lane = (int)(threadIdx.x & 31u);
  int n = n0 + (lane & 15);
  int kbase = (lane < 16) ? 0 : 8;
  const __bf16* p = t + (size_t)n * ld + k0 + kbase;
  v8bf lo = *(const v8bf*)(p);
  v8bf hi = *(const v8bf*)(p + 16);
  return __builtin_shufflevector(lo, hi, 0,1,2,3,4,5,6,7,8,9,10,11,12,13,14,15);
}

// B operand from row-major [K][ld] (strided, scalar gathers).
static __device__ inline v16bf load_B_rm(const __bf16* g, int ld, int k0, int n0) {
  int lane = (int)(threadIdx.x & 31u);
  int n = n0 + (lane & 15);
  int kbase = (lane < 16) ? 0 : 8;
  v16bf r;
#pragma unroll
  for (int j = 0; j < 16; ++j) r[j] = g[(size_t)(k0 + kbase + j + (j >= 8 ? 8 : 0)) * ld + n];
  return r;
}

#if ASYNC_OK
typedef int v4i_g __attribute__((vector_size(16)));
typedef __attribute__((address_space(1))) v4i_g GV4;
typedef __attribute__((address_space(3))) v4i_g LV4;
static __device__ inline void async_copy16(const void* g, void* l) {
  __builtin_amdgcn_global_load_async_to_lds_b128(
      (GV4*)(unsigned long long)g,
      (LV4*)(unsigned int)(unsigned long long)l,
      0, 0);
}
#endif

#if TDM_OK
typedef __attribute__((ext_vector_type(4))) unsigned int v4u;
typedef __attribute__((ext_vector_type(8))) int v8i_t;
typedef __attribute__((ext_vector_type(4))) int v4i_t;
// Pack a Tensor DMA Descriptor (D#) for a 2D (or 1D when tile1==0) bf16 tile
// load Global->LDS, per CDNA5 ISA 8.3/8.4 (group0: count|lds|global|type=2,
// group1: data_size=2B, dims, tile dims, dim0 stride).
static __device__ inline void tdm_load_tile(const void* gsrc, void* ldst,
                                            unsigned tile0, unsigned tile1,
                                            unsigned long long stride0) {
  unsigned long long ga = (unsigned long long)gsrc;
  unsigned la = (unsigned)(unsigned long long)ldst;  // generic LDS ptr low 32 = lds byte addr
  v4u g0;
  g0.x = 1u;                                                  // count=1, user mode
  g0.y = la;                                                  // lds_addr (bytes)
  g0.z = (unsigned)ga;                                        // global_addr[31:0]
  g0.w = (unsigned)((ga >> 32) & 0x01FFFFFFu) | (2u << 30);   // global_addr[56:32] | type=2
  const unsigned long long td0 = 0x7FFFFFFFull;               // huge tensor dims: no OOB clamp
  const unsigned long long td1 = 0x7FFFFFFFull;
  v8i_t g1;
  g1[0] = (int)(1u << 16);                                    // data_size=1 (2 bytes)
  g1[1] = (int)(((unsigned)td0 & 0xFFFFu) << 16);             // tensor_dim0[15:0] @bits63:48
  g1[2] = (int)((((unsigned)(td0 >> 16)) & 0xFFFFu) |         // tensor_dim0[31:16]
                (((unsigned)td1 & 0xFFFFu) << 16));           // tensor_dim1[15:0]
  g1[3] = (int)((((unsigned)(td1 >> 16)) & 0xFFFFu) |         // tensor_dim1[31:16]
                ((tile0 & 0xFFFFu) << 16));                   // tile_dim0
  g1[4] = (int)(tile1 & 0xFFFFu);                             // tile_dim1 (0 => 1D), tile_dim2=0
  g1[5] = (int)(unsigned)(stride0 & 0xFFFFFFFFull);           // tensor_dim0_stride[31:0]
  g1[6] = (int)(unsigned)((stride0 >> 32) & 0xFFFFu);         // dim0_stride[47:32], dim1_stride=0
  g1[7] = 0;
  v4i_t z4 = {0, 0, 0, 0};
#if TDM_SIX
  v8i_t z8 = {0, 0, 0, 0, 0, 0, 0, 0};
  __builtin_amdgcn_tensor_load_to_lds(g0, g1, z4, z4, z8, 0);
#else
  __builtin_amdgcn_tensor_load_to_lds(g0, g1, z4, z4, 0);
#endif
}
#endif

// ---------------- Kernel 1: QKV projection (WMMA over CIN=256) ----------------
// Q stored transposed Qt[b][d][c]; K,V stored [b][c][d]; all bf16.
__global__ __launch_bounds__(128) void qkv_proj_kernel(
    const float* __restrict__ x,
    const float* __restrict__ wq, const float* __restrict__ bq,
    const float* __restrict__ wk, const float* __restrict__ bk,
    const float* __restrict__ wv, const float* __restrict__ bv,
    __bf16* __restrict__ Qt, __bf16* __restrict__ Kg, __bf16* __restrict__ Vg) {
  __shared__ __attribute__((aligned(32))) __bf16 xs[64 * CIN];  // [d][f] transposed, 32KB
  int b = blockIdx.y;
  int d0 = blockIdx.x * 64;
  int tid = (int)threadIdx.x;
  const float* xb = x + (size_t)b * CIN * HWD;
  for (int i = tid; i < CIN * 16; i += 128) {
    int f = i >> 4;
    int d4 = (i & 15) * 4;
    float4 v = *(const float4*)(xb + (size_t)f * HWD + d0 + d4);
    xs[(d4 + 0) * CIN + f] = (__bf16)v.x;
    xs[(d4 + 1) * CIN + f] = (__bf16)v.y;
    xs[(d4 + 2) * CIN + f] = (__bf16)v.z;
    xs[(d4 + 3) * CIN + f] = (__bf16)v.w;
  }
  __syncthreads();
  int w = tid >> 5;
  int lane = tid & 31;
  int nlane = lane & 15;
  int off = (lane < 16) ? 0 : 8;
  for (int u = 0; u < 3; ++u) {
    int unit = w * 3 + u;                 // 12 units: 3 projections x 4 row tiles
    int p = unit >> 2, ct = unit & 3;
    const float* W  = (p == 0) ? wq : (p == 1) ? wk : wv;
    const float* Bs = (p == 0) ? bq : (p == 1) ? bk : bv;
    v8f acc[4];
#pragma unroll
    for (int et = 0; et < 4; ++et) acc[et] = vzero8();
    for (int kb = 0; kb < 8; ++kb) {
      v16bf a = load_A_f32(W, CIN, ct * 16, kb * 32);
#pragma unroll
      for (int et = 0; et < 4; ++et) {
        v16bf bm = load_B_tr(xs, CIN, et * 16, kb * 32);
        acc[et] = wmma_bf16(a, bm, acc[et]);
      }
    }
    float bias8[8];
#pragma unroll
    for (int i = 0; i < 8; ++i) bias8[i] = Bs[ct * 16 + off + i];
    if (p == 0) {  // Qt[d][c]: rows (c) contiguous per lane -> packed 16B store
#pragma unroll
      for (int et = 0; et < 4; ++et) {
        int d = d0 + et * 16 + nlane;
        v8bf pk;
#pragma unroll
        for (int i = 0; i < 8; ++i) pk[i] = (__bf16)(acc[et][i] + bias8[i]);
        *(v8bf*)(Qt + ((size_t)b * HWD + d) * CQ + ct * 16 + off) = pk;
      }
    } else {       // K/V [c][d]
      __bf16* O = (p == 1) ? Kg : Vg;
#pragma unroll
      for (int et = 0; et < 4; ++et) {
        int d = d0 + et * 16 + nlane;
#pragma unroll
        for (int i = 0; i < 8; ++i) {
          int c = ct * 16 + off + i;
          O[((size_t)b * CQ + c) * HWD + d] = (__bf16)(acc[et][i] + bias8[i]);
        }
      }
    }
  }
}

// ---------------- Kernel 2: softmax row stats m[d], l[d] (online, WMMA) ------
__global__ __launch_bounds__(32) void softmax_stats_kernel(
    const __bf16* __restrict__ Qt, const __bf16* __restrict__ Kg,
    float* __restrict__ Mr, float* __restrict__ Lr) {
  int b = blockIdx.y;
  int d0 = blockIdx.x * 64;
  const __bf16* Qtb = Qt + (size_t)b * HWD * CQ;
  const __bf16* Kb  = Kg + (size_t)b * CQ * HWD;
  v16bf aq[4][2];
#pragma unroll
  for (int dt = 0; dt < 4; ++dt)
#pragma unroll
    for (int kb = 0; kb < 2; ++kb)
      aq[dt][kb] = load_A_rm(Qtb, CQ, d0 + dt * 16, kb * 32);
  float rm[4][8], rl[4][8];
#pragma unroll
  for (int dt = 0; dt < 4; ++dt)
#pragma unroll
    for (int i = 0; i < 8; ++i) { rm[dt][i] = -3.0e38f; rl[dt][i] = 0.f; }
  for (int e16 = 0; e16 < HWD / 16; ++e16) {
    v16bf b0 = load_B_rm(Kb, HWD, 0,  e16 * 16);
    v16bf b1 = load_B_rm(Kb, HWD, 32, e16 * 16);
#pragma unroll
    for (int dt = 0; dt < 4; ++dt) {
      v8f s = vzero8();
      s = wmma_bf16(aq[dt][0], b0, s);
      s = wmma_bf16(aq[dt][1], b1, s);
#pragma unroll
      for (int i = 0; i < 8; ++i) {
        float v = s[i];
        float tm = v;  // row max over the 16 lanes of this half-wave
        tm = fmaxf(tm, __shfl_xor(tm, 1));
        tm = fmaxf(tm, __shfl_xor(tm, 2));
        tm = fmaxf(tm, __shfl_xor(tm, 4));
        tm = fmaxf(tm, __shfl_xor(tm, 8));
        float nm = fmaxf(rm[dt][i], tm);
        float p = __expf(v - nm);
        p += __shfl_xor(p, 1);
        p += __shfl_xor(p, 2);
        p += __shfl_xor(p, 4);
        p += __shfl_xor(p, 8);
        rl[dt][i] = rl[dt][i] * __expf(rm[dt][i] - nm) + p;
        rm[dt][i] = nm;
      }
    }
  }
  int lane = (int)(threadIdx.x & 31u);
  int off = (lane < 16) ? 0 : 8;
  if ((lane & 15) == 0) {
#pragma unroll
    for (int dt = 0; dt < 4; ++dt)
#pragma unroll
      for (int i = 0; i < 8; ++i) {
        int d = d0 + dt * 16 + off + i;
        Mr[(size_t)b * HWD + d] = rm[dt][i];
        Lr[(size_t)b * HWD + d] = rl[dt][i];
      }
  }
}

// ---------------- Kernel 3: fused S-recompute + P + O = V*P ------------------
__global__ __launch_bounds__(64) void attn_fused_kernel(
    const __bf16* __restrict__ Qt, const __bf16* __restrict__ Kg,
    const __bf16* __restrict__ Vg, const float* __restrict__ Mr,
    const float* __restrict__ Lr, __bf16* __restrict__ Ob) {
  __shared__ __attribute__((aligned(32))) __bf16 KtS[128 * 64];   // [e_local][c] 16KB
  __shared__ __attribute__((aligned(32))) __bf16 QtS[64 * 64];    // [d][c]       8KB
  __shared__ __attribute__((aligned(32))) __bf16 VcS[64 * 64];    // [c][d]       8KB
  __shared__ __attribute__((aligned(32))) __bf16 PS[2][64 * 64];  // per-wave [n][d] 16KB
  int b = blockIdx.y;
  int eWG = blockIdx.x * 128;
  int tid = (int)threadIdx.x;
  int w = tid >> 5;
  int lane = tid & 31;
  int nlane = lane & 15;
  int off = (lane < 16) ? 0 : 8;
  const __bf16* Qtb = Qt + (size_t)b * HWD * CQ;
  const __bf16* Kb  = Kg + (size_t)b * CQ * HWD;
  const __bf16* Vb  = Vg + (size_t)b * CQ * HWD;
  const float* mb = Mr + (size_t)b * HWD;
  const float* lb = Lr + (size_t)b * HWD;
  int e0l = w * 64;

  // Stage K slab transposed [e][c] once (coalesced global reads).
  for (int i = tid; i < 64 * 16; i += 64) {
    int c = i >> 4;
    int e8 = (i & 15) * 8;
    v8bf v = *(const v8bf*)(Kb + (size_t)c * HWD + eWG + e8);
#pragma unroll
    for (int j = 0; j < 8; ++j) KtS[(e8 + j) * 64 + c] = v[j];
  }
  __syncthreads();

  v8f acc[4][4];
#pragma unroll
  for (int ct = 0; ct < 4; ++ct)
#pragma unroll
    for (int et = 0; et < 4; ++et) acc[ct][et] = vzero8();

  for (int ec = 0; ec < HWD / 64; ++ec) {
    int dd0 = ec * 64;
    __syncthreads();  // protect QtS/VcS reuse
#if TDM_OK
    // Tensor Data Mover: wave0 DMAs the contiguous Q chunk (1D tile),
    // wave1 DMAs the strided 64x64 V tile; each waits its own TENSORcnt.
    if (w == 0) {
      tdm_load_tile(Qtb + (size_t)dd0 * CQ, QtS, 4096u, 0u, 4096ull);
    } else {
      tdm_load_tile(Vb + dd0, VcS, 64u, 64u, (unsigned long long)HWD);
    }
    __builtin_amdgcn_s_wait_tensorcnt(0);
#elif ASYNC_OK
    for (int i = tid; i < 512; i += 64)
      async_copy16(Qtb + (size_t)dd0 * CQ + i * 8, QtS + i * 8);
    for (int i = tid; i < 512; i += 64) {
      int c = i >> 3, part = i & 7;
      async_copy16(Vb + (size_t)c * HWD + dd0 + part * 8, VcS + c * 64 + part * 8);
    }
    __builtin_amdgcn_s_wait_asynccnt(0);
#else
    for (int i = tid; i < 512; i += 64)
      *(uint4*)(QtS + i * 8) = *(const uint4*)(Qtb + (size_t)dd0 * CQ + i * 8);
    for (int i = tid; i < 512; i += 64) {
      int c = i >> 3, part = i & 7;
      *(uint4*)(VcS + c * 64 + part * 8) =
          *(const uint4*)(Vb + (size_t)c * HWD + dd0 + part * 8);
    }
#endif
    __syncthreads();

    // S tiles + P = exp(S - m[d]) / l[d], stored transposed PS[w][n][d]
#pragma unroll
    for (int dt = 0; dt < 4; ++dt) {
      float mr[8], li[8];
#pragma unroll
      for (int i = 0; i < 8; ++i) {
        int d = dd0 + dt * 16 + off + i;
        mr[i] = mb[d];
        li[i] = 1.0f / lb[d];
      }
      v16bf a0 = load_A_rm(QtS, 64, dt * 16, 0);
      v16bf a1 = load_A_rm(QtS, 64, dt * 16, 32);
#pragma unroll
      for (int et = 0; et < 4; ++et) {
        v8f s = vzero8();
        s = wmma_bf16(a0, load_B_tr(KtS, 64, e0l + et * 16, 0),  s);
        s = wmma_bf16(a1, load_B_tr(KtS, 64, e0l + et * 16, 32), s);
        v8bf pk;
#pragma unroll
        for (int i = 0; i < 8; ++i)
          pk[i] = (__bf16)(__expf(s[i] - mr[i]) * li[i]);
        *(v8bf*)(&PS[w][(size_t)(et * 16 + nlane) * 64 + dt * 16 + off]) = pk;
      }
    }

    // O += V[c][d] * P[d][e]
#pragma unroll
    for (int kb = 0; kb < 2; ++kb) {
      v16bf bp[4];
#pragma unroll
      for (int et = 0; et < 4; ++et)
        bp[et] = load_B_tr(&PS[w][0], 64, et * 16, kb * 32);
#pragma unroll
      for (int ct = 0; ct < 4; ++ct) {
        v16bf av = load_A_rm(VcS, 64, ct * 16, kb * 32);
#pragma unroll
        for (int et = 0; et < 4; ++et)
          acc[ct][et] = wmma_bf16(av, bp[et], acc[ct][et]);
      }
    }
  }

  // Store O as bf16 [c][e]
#pragma unroll
  for (int ct = 0; ct < 4; ++ct)
#pragma unroll
    for (int et = 0; et < 4; ++et)
#pragma unroll
      for (int i = 0; i < 8; ++i) {
        int c = ct * 16 + off + i;
        int e = eWG + e0l + et * 16 + nlane;
        Ob[((size_t)b * CQ + c) * HWD + e] = (__bf16)acc[ct][et][i];
      }
}

// ---------------- Kernel 4: output projection [256x64]@[64x4096] + bias ------
__global__ __launch_bounds__(32) void out_proj_kernel(
    const __bf16* __restrict__ Ob, const float* __restrict__ wo,
    const float* __restrict__ bo, float* __restrict__ out) {
  int b = blockIdx.y;
  int e0 = blockIdx.x * 64;
  const __bf16* Obb = Ob + (size_t)b * CQ * HWD;
  float* outb = out + (size_t)b * CIN * HWD;
  int lane = (int)(threadIdx.x & 31u);
  int nlane = lane & 15;
  int off = (lane < 16) ? 0 : 8;
  v16bf bmat[4][2];
#pragma unroll
  for (int et = 0; et < 4; ++et)
#pragma unroll
    for (int kb = 0; kb < 2; ++kb)
      bmat[et][kb] = load_B_rm(Obb, HWD, kb * 32, e0 + et * 16);
  for (int ct = 0; ct < 16; ++ct) {
    v16bf a0 = load_A_f32(wo, CQ, ct * 16, 0);
    v16bf a1 = load_A_f32(wo, CQ, ct * 16, 32);
    float bias8[8];
#pragma unroll
    for (int i = 0; i < 8; ++i) bias8[i] = bo[ct * 16 + off + i];
#pragma unroll
    for (int et = 0; et < 4; ++et) {
      v8f a = vzero8();
      a = wmma_bf16(a0, bmat[et][0], a);
      a = wmma_bf16(a1, bmat[et][1], a);
#pragma unroll
      for (int i = 0; i < 8; ++i) {
        int f = ct * 16 + off + i;
        outb[(size_t)f * HWD + e0 + et * 16 + nlane] = a[i] + bias8[i];
      }
    }
  }
}

extern "C" void kernel_launch(void* const* d_in, const int* in_sizes, int n_in,
                              void* d_out, int out_size, void* d_ws, size_t ws_size,
                              hipStream_t stream) {
  (void)in_sizes; (void)n_in; (void)out_size; (void)ws_size;
  const float* x  = (const float*)d_in[0];
  const float* wq = (const float*)d_in[1];
  const float* bq = (const float*)d_in[2];
  const float* wk = (const float*)d_in[3];
  const float* bk = (const float*)d_in[4];
  const float* wv = (const float*)d_in[5];
  const float* bv = (const float*)d_in[6];
  const float* wo = (const float*)d_in[7];
  const float* bo = (const float*)d_in[8];
  float* out = (float*)d_out;

  const size_t nMat = (size_t)NB * HWD * CQ;  // 1M elems per bf16 matrix
  __bf16* Qt = (__bf16*)d_ws;     // [b][d][c]
  __bf16* Kg = Qt + nMat;         // [b][c][d]
  __bf16* Vg = Kg + nMat;         // [b][c][d]
  __bf16* Ob = Vg + nMat;         // [b][c][d]
  float*  Mr = (float*)(Ob + nMat);      // [b][d]
  float*  Lr = Mr + (size_t)NB * HWD;    // [b][d]

  qkv_proj_kernel<<<dim3(HWD / 64, NB), 128, 0, stream>>>(
      x, wq, bq, wk, bk, wv, bv, Qt, Kg, Vg);
  softmax_stats_kernel<<<dim3(HWD / 64, NB), 32, 0, stream>>>(Qt, Kg, Mr, Lr);
  attn_fused_kernel<<<dim3(HWD / 128, NB), 64, 0, stream>>>(Qt, Kg, Vg, Mr, Lr, Ob);
  out_proj_kernel<<<dim3(HWD / 64, NB), 32, 0, stream>>>(Ob, wo, bo, out);
}
